// ComplexAutoencoder_1717986919076
// MI455X (gfx1250) — matrix operand order repaired and checked by reading, not compiled, not run on hardware
//
#include <hip/hip_runtime.h>
#include <cstdint>
#include <cstddef>

typedef __attribute__((ext_vector_type(16))) _Float16 v16h;
typedef __attribute__((ext_vector_type(8)))  _Float16 v8h;
typedef __attribute__((ext_vector_type(8)))  float    v8f;

#define T_STEPS 256
#define NB      8
#define NF      64
#define ROWS    (T_STEPS * NB)   // 2048
#define NW      8                // waves in the scan workgroup (wave32 -> 256 threads)
#define MAXDH   256

// ---------------------------------------------------------------------------
// Complex scalar helpers (f32)
// ---------------------------------------------------------------------------
__device__ __forceinline__ float2 cmul(float2 a, float2 b) {
  return make_float2(a.x * b.x - a.y * b.y, a.x * b.y + a.y * b.x);
}
__device__ __forceinline__ float2 cexp(float2 z) {
  float e = __expf(z.x);
  float s, c;
  __sincosf(z.y, &s, &c);
  return make_float2(e * c, e * s);
}
__device__ __forceinline__ float2 cinv(float2 z) {
  float d = 1.0f / (z.x * z.x + z.y * z.y);
  return make_float2(z.x * d, -z.y * d);
}
__device__ __forceinline__ float2 csig(float2 z) {
  // 1 / (1 + exp(-z))
  float2 w = cexp(make_float2(-z.x, -z.y));
  return cinv(make_float2(1.0f + w.x, w.y));
}
__device__ __forceinline__ float2 ctanhc(float2 z) {
  // (e^{2z} - 1) / (e^{2z} + 1)
  float2 u = cexp(make_float2(2.0f * z.x, 2.0f * z.y));
  return cmul(make_float2(u.x - 1.0f, u.y), cinv(make_float2(u.x + 1.0f, u.y)));
}

// ---------------------------------------------------------------------------
// WMMA fragment loaders (wave32; this code's contiguous k = 16*half + s order, same for A and B).
// With split re/im f16 planes, every fragment is two contiguous 16B chunks
// per lane -> 2x b128 (global) or 2x ds_load_b128 (LDS):
//   A (16x32): lane m=lane&15, half=lane>>4:
//     frag[0..15] = plane[m*ld + kbase + 16*half .. +15]
//   B (32x16, from W stored [G,K] row-major, B[k][n] = w[(gbase+n)*ld + k]):
//     frag[0..15] = w[(gbase+n)*ld + kbase + 16*half .. +15]
//   C/D: lane%16 = N, VGPR r -> M = r + 8*half
// ---------------------------------------------------------------------------
__device__ __forceinline__ v16h combine8(v8h lo, v8h hi) {
  return __builtin_shufflevector(lo, hi, 0, 1, 2, 3, 4, 5, 6, 7,
                                         8, 9, 10, 11, 12, 13, 14, 15);
}

__device__ __forceinline__ v16h load_a_plane(const _Float16* p, int ld,
                                             int rbase, int kbase) {
  int lane = threadIdx.x & 31;
  int m = lane & 15, half = lane >> 4;
  const _Float16* q = p + (size_t)(rbase + m) * ld + kbase + 16 * half;
  v8h lo = *(const v8h*)(q);
  v8h hi = *(const v8h*)(q + 8);
  return combine8(lo, hi);
}

__device__ __forceinline__ v16h load_b_plane(const _Float16* __restrict__ w, int ld,
                                             int gbase, int kbase) {
  int lane = threadIdx.x & 31;
  int n = lane & 15, half = lane >> 4;
  const _Float16* q = w + (size_t)(gbase + n) * ld + kbase + 16 * half;
  v8h lo = *(const v8h*)(q);
  v8h hi = *(const v8h*)(q + 8);
  return combine8(lo, hi);
}

// Elementwise negate a f16 fragment (packed VALU; f16 WMMA has no A/B neg).
__device__ __forceinline__ v16h vneg16(v16h a) {
  v16h r;
#pragma unroll
  for (int i = 0; i < 16; ++i) r[i] = -a[i];
  return r;
}

// Complex MMA: (Ar + iAi)(Br + iBi) accumulated into (Cr, Ci).
// ain must hold -Ai (precomputed once per K-chunk, shared across tiles).
__device__ __forceinline__ void cwmma(const v16h& ar, const v16h& ai, const v16h& ain,
                                      const v16h& br, const v16h& bi,
                                      v8f& cr, v8f& ci) {
  cr = __builtin_amdgcn_wmma_f32_16x16x32_f16(false, ar,  false, br, (short)0, cr, false, false);
  cr = __builtin_amdgcn_wmma_f32_16x16x32_f16(false, ain, false, bi, (short)0, cr, false, false);
  ci = __builtin_amdgcn_wmma_f32_16x16x32_f16(false, ar,  false, bi, (short)0, ci, false, false);
  ci = __builtin_amdgcn_wmma_f32_16x16x32_f16(false, ai,  false, br, (short)0, ci, false, false);
}

// ---------------------------------------------------------------------------
// Small data-movement kernels
// ---------------------------------------------------------------------------
// Split complex f32 weights into re/im f16 planes (same [G,K] layout).
__global__ void convert_w_kernel(const float2* __restrict__ w,
                                 _Float16* __restrict__ wre,
                                 _Float16* __restrict__ wim, int n) {
  int i = blockIdx.x * blockDim.x + threadIdx.x;
  if (i < n) {
    float2 v = w[i];
    wre[i] = (_Float16)v.x;
    wim[i] = (_Float16)v.y;
  }
}

// x: [B, T, F] complex f32 -> time-major f16 planes [(t*NB+b), F]
__global__ void pack_x_kernel(const float2* __restrict__ x,
                              _Float16* __restrict__ aRe,
                              _Float16* __restrict__ aIm) {
  int i = blockIdx.x * blockDim.x + threadIdx.x;
  if (i >= ROWS * NF) return;
  int f = i % NF;
  int rb = i / NF;
  int b = rb % NB, t = rb / NB;
  float2 v = x[((size_t)b * T_STEPS + t) * NF + f];
  aRe[i] = (_Float16)v.x;
  aIm[i] = (_Float16)v.y;
}

// ytmp: [(t*NB+b), F] complex f32 -> out: [B, T, F] complex f32
__global__ void unpack_y_kernel(const float2* __restrict__ ytmp, float2* __restrict__ out) {
  int i = blockIdx.x * blockDim.x + threadIdx.x;
  if (i >= ROWS * NF) return;
  int f = i % NF;
  int bt = i / NF;
  int t = bt % T_STEPS, b = bt / T_STEPS;
  out[i] = ytmp[((size_t)(t * NB + b)) * NF + f];
}

// ---------------------------------------------------------------------------
// px GEMM: px[row, g] = sum_k act[row, k] * wih[g, k] + bias[g]  (complex)
// One wave per 16x16 output tile; activations and weights are f16 planes.
// ---------------------------------------------------------------------------
template <int DIN>
__global__ __launch_bounds__(32)
void gemm_px_kernel(const _Float16* __restrict__ aRe,
                    const _Float16* __restrict__ aIm,
                    const _Float16* __restrict__ wre,
                    const _Float16* __restrict__ wim,
                    const float2* __restrict__ bias,
                    float2* __restrict__ px, int g4) {
  int rbase = blockIdx.x * 16;
  int gbase = blockIdx.y * 16;
  v8f cr = {}, ci = {};
#pragma unroll
  for (int k = 0; k < DIN; k += 32) {
    v16h ar = load_a_plane(aRe, DIN, rbase, k);
    v16h ai = load_a_plane(aIm, DIN, rbase, k);
    v16h ain = vneg16(ai);
    v16h br = load_b_plane(wre, DIN, gbase, k);
    v16h bi = load_b_plane(wim, DIN, gbase, k);
    cwmma(ar, ai, ain, br, bi, cr, ci);
  }
  int lane = threadIdx.x & 31;
  int n = lane & 15, half = lane >> 4;
#pragma unroll
  for (int r = 0; r < 8; ++r) {
    int row = rbase + r + 8 * half;
    int g = gbase + n;
    float2 bb = bias[g];
    px[(size_t)row * g4 + g] = make_float2(cr[r] + bb.x, ci[r] + bb.y);
  }
}

// ---------------------------------------------------------------------------
// Persistent LSTM scan: one workgroup (8 waves), T=256 serial steps.
//  - h kept as f16 re/im planes in LDS (16 rows: batch 0..7 + zero pad 8..15)
//  - recurrent GEMM [16 x DH] x [DH x 4DH] via complex WMMA (ds_load_b128 A)
//  - gate math fully in registers: each lane of a j-tile owns i,f,g,o for
//    8 (batch, j) pairs; cell state c lives in VGPRs across all 256 steps.
//  - h written out as f16 planes = next layer's GEMM A operand (no re-convert)
// ---------------------------------------------------------------------------
template <int DH>
__global__ __launch_bounds__(NW * 32, 1)
void lstm_scan_kernel(const float2* __restrict__ px,
                      const _Float16* __restrict__ wre,
                      const _Float16* __restrict__ wim,
                      _Float16* __restrict__ houtRe,
                      _Float16* __restrict__ houtIm) {
  constexpr int G4 = 4 * DH;
  constexpr int NJT = DH / 16;                  // j tiles over hidden dim
  constexpr int MAXJT = (NJT + NW - 1) / NW;    // tiles per wave

  __shared__ _Float16 s_hre[16][MAXDH];
  __shared__ _Float16 s_him[16][MAXDH];

  int tid = threadIdx.x;
  int wave = tid >> 5, lane = tid & 31;
  int n = lane & 15, half = lane >> 4;

  // zero hidden state (incl. pad rows 8..15 which stay zero forever)
  for (int i = tid; i < 16 * MAXDH; i += NW * 32) {
    (&s_hre[0][0])[i] = (_Float16)0.0f;
    (&s_him[0][0])[i] = (_Float16)0.0f;
  }

  // cell state in registers (meaningful in lanes 0..15 only)
  float cR[MAXJT][8], cI[MAXJT][8];
#pragma unroll
  for (int u = 0; u < MAXJT; ++u)
#pragma unroll
    for (int r = 0; r < 8; ++r) { cR[u][r] = 0.0f; cI[u][r] = 0.0f; }

  __syncthreads();

  for (int t = 0; t < T_STEPS; ++t) {
    if (t + 1 < T_STEPS)  // pull next step's gate preactivations toward the WGP
      __builtin_prefetch(px + (size_t)((t + 1) * NB) * G4, 0, 1);

    // ---- phase 1: recurrent complex GEMM (reads h from LDS) ----
    v8f accR[MAXJT][4], accI[MAXJT][4];
    v8f vzero = {};
#pragma unroll
    for (int u = 0; u < MAXJT; ++u)
#pragma unroll
      for (int g = 0; g < 4; ++g) { accR[u][g] = vzero; accI[u][g] = vzero; }

    for (int k = 0; k < DH; k += 32) {
      v16h ar = load_a_plane(&s_hre[0][0], MAXDH, 0, k);
      v16h ai = load_a_plane(&s_him[0][0], MAXDH, 0, k);
      v16h ain = vneg16(ai);
#pragma unroll
      for (int u = 0; u < MAXJT; ++u) {
        int jt = wave + u * NW;
        if (jt < NJT) {
#pragma unroll
          for (int g = 0; g < 4; ++g) {
            v16h br = load_b_plane(wre, DH, g * DH + jt * 16, k);
            v16h bi = load_b_plane(wim, DH, g * DH + jt * 16, k);
            cwmma(ar, ai, ain, br, bi, accR[u][g], accI[u][g]);
          }
        }
      }
    }
    __syncthreads();  // all reads of h(t) done before anyone writes h(t+1)

    // ---- phase 2: gate nonlinearity + state update (registers only) ----
#pragma unroll
    for (int u = 0; u < MAXJT; ++u) {
      int jt = wave + u * NW;
      if (jt < NJT && half == 0) {    // lanes 0..15 hold batches 0..7 (M=r)
        int j = jt * 16 + n;
#pragma unroll
        for (int r = 0; r < 8; ++r) {
          int b = r;
          const float2* prow = px + (size_t)(t * NB + b) * G4;
          float2 gi = prow[0 * DH + j];
          float2 gf = prow[1 * DH + j];
          float2 gg = prow[2 * DH + j];
          float2 go = prow[3 * DH + j];
          gi.x += accR[u][0][r];  gi.y += accI[u][0][r];
          gf.x += accR[u][1][r];  gf.y += accI[u][1][r];
          gg.x += accR[u][2][r];  gg.y += accI[u][2][r];
          go.x += accR[u][3][r];  go.y += accI[u][3][r];

          float2 si = csig(gi), sf = csig(gf), so = csig(go);
          float2 tg = ctanhc(gg);
          float2 cc = cmul(sf, make_float2(cR[u][r], cI[u][r]));
          float2 it = cmul(si, tg);
          cc.x += it.x; cc.y += it.y;
          cR[u][r] = cc.x; cI[u][r] = cc.y;

          float2 h = cmul(so, ctanhc(cc));
          _Float16 hr = (_Float16)h.x;
          _Float16 hi = (_Float16)h.y;
          s_hre[b][j] = hr;
          s_him[b][j] = hi;
          size_t o = (size_t)(t * NB + b) * DH + j;
          houtRe[o] = hr;
          houtIm[o] = hi;
        }
      }
    }
    __syncthreads();  // h(t+1) visible to all waves before next step
  }
}

// ---------------------------------------------------------------------------
// Host-side orchestration (stream-ordered only; scratch in d_ws)
// ---------------------------------------------------------------------------
static void launch_gemm(int din, const _Float16* aRe, const _Float16* aIm,
                        const _Float16* wre, const _Float16* wim,
                        const float2* bias, float2* px, int g4, hipStream_t s) {
  dim3 grid(ROWS / 16, g4 / 16);
  switch (din) {
    case 64:  gemm_px_kernel<64><<<grid, 32, 0, s>>>(aRe, aIm, wre, wim, bias, px, g4);  break;
    case 128: gemm_px_kernel<128><<<grid, 32, 0, s>>>(aRe, aIm, wre, wim, bias, px, g4); break;
    case 256: gemm_px_kernel<256><<<grid, 32, 0, s>>>(aRe, aIm, wre, wim, bias, px, g4); break;
    default: break;
  }
}

static void launch_scan(int dh, const float2* px, const _Float16* wre,
                        const _Float16* wim, _Float16* hRe, _Float16* hIm,
                        hipStream_t s) {
  switch (dh) {
    case 64:  lstm_scan_kernel<64><<<1, NW * 32, 0, s>>>(px, wre, wim, hRe, hIm);  break;
    case 128: lstm_scan_kernel<128><<<1, NW * 32, 0, s>>>(px, wre, wim, hRe, hIm); break;
    case 256: lstm_scan_kernel<256><<<1, NW * 32, 0, s>>>(px, wre, wim, hRe, hIm); break;
    default: break;
  }
}

extern "C" void kernel_launch(void* const* d_in, const int* in_sizes, int n_in,
                              void* d_out, int out_size, void* d_ws, size_t ws_size,
                              hipStream_t stream) {
  (void)in_sizes; (void)n_in; (void)out_size; (void)ws_size;

  const float2* x = (const float2*)d_in[0];
  struct Stack { const float2 *wih0, *wihr, *whh, *b; int din, dh; };
  Stack st[4] = {
    { (const float2*)d_in[1],  (const float2*)d_in[2],  (const float2*)d_in[3],  (const float2*)d_in[4],  64, 256 },
    { (const float2*)d_in[5],  (const float2*)d_in[6],  (const float2*)d_in[7],  (const float2*)d_in[8],  256, 128 },
    { (const float2*)d_in[9],  (const float2*)d_in[10], (const float2*)d_in[11], (const float2*)d_in[12], 128, 256 },
    { (const float2*)d_in[13], (const float2*)d_in[14], (const float2*)d_in[15], (const float2*)d_in[16], 256, 64 },
  };
  const float2* lin_w = (const float2*)d_in[17];
  const float2* lin_b = (const float2*)d_in[18];

  // workspace carve-up (all regions 16B-aligned by construction)
  char* ws = (char*)d_ws;
  _Float16* actReA = (_Float16*)ws; ws += (size_t)ROWS * MAXDH * sizeof(_Float16);  // 1 MB
  _Float16* actImA = (_Float16*)ws; ws += (size_t)ROWS * MAXDH * sizeof(_Float16);
  _Float16* actReB = (_Float16*)ws; ws += (size_t)ROWS * MAXDH * sizeof(_Float16);
  _Float16* actImB = (_Float16*)ws; ws += (size_t)ROWS * MAXDH * sizeof(_Float16);
  float2*   px     = (float2*)ws;   ws += (size_t)ROWS * 4 * MAXDH * sizeof(float2); // 16 MB
  _Float16* wreP = (_Float16*)ws; ws += (size_t)4 * MAXDH * MAXDH * sizeof(_Float16);
  _Float16* wimP = (_Float16*)ws; ws += (size_t)4 * MAXDH * MAXDH * sizeof(_Float16);
  _Float16* hreP = (_Float16*)ws; ws += (size_t)4 * MAXDH * MAXDH * sizeof(_Float16);
  _Float16* himP = (_Float16*)ws; ws += (size_t)4 * MAXDH * MAXDH * sizeof(_Float16);

  pack_x_kernel<<<(ROWS * NF + 255) / 256, 256, 0, stream>>>(x, actReA, actImA);

  _Float16* curRe = actReA; _Float16* curIm = actImA;
  _Float16* nxtRe = actReB; _Float16* nxtIm = actImB;
  for (int s = 0; s < 4; ++s) {
    int dh = st[s].dh, g4 = 4 * dh;
    for (int l = 0; l < 2; ++l) {
      int din = (l == 0) ? st[s].din : dh;
      const float2* wih  = (l == 0) ? st[s].wih0 : st[s].wihr;  // wihr has L-1==1 slice
      const float2* whh  = st[s].whh + (size_t)l * g4 * dh;
      const float2* bias = st[s].b   + (size_t)l * g4;

      int nwih = g4 * din;
      convert_w_kernel<<<(nwih + 255) / 256, 256, 0, stream>>>(wih, wreP, wimP, nwih);
      launch_gemm(din, curRe, curIm, wreP, wimP, bias, px, g4, stream);

      int nwhh = g4 * dh;
      convert_w_kernel<<<(nwhh + 255) / 256, 256, 0, stream>>>(whh, hreP, himP, nwhh);
      launch_scan(dh, px, hreP, himP, nxtRe, nxtIm, stream);

      _Float16* tr = curRe; curRe = nxtRe; nxtRe = tr;
      _Float16* ti = curIm; curIm = nxtIm; nxtIm = ti;
    }
  }

  // final complex linear: y = h @ W^T + b  (reuses px as [ROWS, 64] scratch)
  convert_w_kernel<<<(NF * NF + 255) / 256, 256, 0, stream>>>(lin_w, wreP, wimP, NF * NF);
  launch_gemm(NF, curRe, curIm, wreP, wimP, lin_b, px, NF, stream);
  unpack_y_kernel<<<(ROWS * NF + 255) / 256, 256, 0, stream>>>(px, (float2*)d_out);
}
